// Attention_47768626266691
// MI455X (gfx1250) — compile-verified, matrix operand order checked by reading, not run on hardware
//
#include <hip/hip_runtime.h>

// ---------------------------------------------------------------------------
// Problem constants (from reference): B=2, S=2048, D=1024, HQ=16, HKV=4, HD=64
// ---------------------------------------------------------------------------
constexpr int Bc   = 2;
constexpr int Sc   = 2048;
constexpr int Dc   = 1024;
constexpr int HQc  = 16;
constexpr int HKVc = 4;
constexpr int HDc  = 64;
constexpr int MTOT = Bc * Sc;              // 4096 rows for all projections
#define SCALE_F 0.125f                     // 1/sqrt(64)

typedef __attribute__((ext_vector_type(16))) __bf16 v16bf;
typedef __attribute__((ext_vector_type(8)))  __bf16 v8bf;
typedef __attribute__((ext_vector_type(8)))  float  v8f;

union Frag {
    v16bf v;
    v8bf  h[2];
};

// reduce across the 16-lane half (rows live per half on CDNA5 C-frags)
__device__ __forceinline__ float redmax16(float v) {
#pragma unroll
    for (int m = 1; m <= 8; m <<= 1) v = fmaxf(v, __shfl_xor(v, m, 32));
    return v;
}
__device__ __forceinline__ float redsum16(float v) {
#pragma unroll
    for (int m = 1; m <= 8; m <<= 1) v += __shfl_xor(v, m, 32);
    return v;
}

// ---------------------------------------------------------------------------
// fp32 -> bf16 elementwise convert
// ---------------------------------------------------------------------------
__global__ void cvt_bf16_kernel(const float* __restrict__ in, __bf16* __restrict__ out, int n) {
    int i = blockIdx.x * blockDim.x + threadIdx.x;
    if (i < n) out[i] = (__bf16)in[i];
}

// W (K x N, row-major fp32)  ->  WT (N x K, row-major bf16)
__global__ void transpose_cvt_kernel(const float* __restrict__ W, __bf16* __restrict__ WT,
                                     int K, int N) {
    int idx = blockIdx.x * blockDim.x + threadIdx.x;
    if (idx >= K * N) return;
    int k = idx % K;            // fastest -> coalesced writes
    int n = idx / K;
    WT[(size_t)n * K + k] = (__bf16)W[(size_t)k * N + n];
}

// ---------------------------------------------------------------------------
// Generic bf16 WMMA GEMM:  C(MxN,f32) = A(MxK,bf16 row-major) * Bt(NxK)^T + bias
// Wave computes a 16x64 tile; 8 waves/block -> 128 rows x 64 cols per block.
// All 8 waves share the same 64 output columns, so the shared Bt tile
// (64 rows x 32 k = 4KB bf16) is staged block-cooperatively into LDS via the
// gfx1250 async-copy path (global_load_async_to_lds_b128 / ASYNCcnt),
// double-buffered against the WMMA consumption of the previous tile.
// grid = (N/64, M/128), block = 256.  Requires M%128==0, N%64==0, K%32==0.
// ---------------------------------------------------------------------------
__global__ void gemm_bias_bf16_kernel(const __bf16* __restrict__ A,
                                      const __bf16* __restrict__ Bt,
                                      const float* __restrict__ bias,
                                      float* __restrict__ C,
                                      int M, int N, int K) {
    __shared__ __align__(16) __bf16 ldsB[2][64][32];   // double-buffered B tile (8 KB)

    const int tid  = threadIdx.x;
    const int wave = tid >> 5;
    const int lane = tid & 31;
    const int lh   = lane & 15;
    const int sel  = lane >> 4;
    const int m0   = blockIdx.y * 128 + wave * 16;
    const int n0   = blockIdx.x * 64;

    // staging assignment: 256 threads x 16B = one 64x32 bf16 tile per issue
    const int srow = tid >> 2;          // 0..63  (column of the GEMM / row of Bt)
    const int schk = (tid & 3) * 8;     // 16-byte chunk within the 32-element k-slice
    const __bf16* sgbase = Bt + (size_t)(n0 + srow) * K + schk;

    auto stage = [&](int kt, int buf) {
        uint64_t ga = (uint64_t)(uintptr_t)(sgbase + kt * 32);
        uint32_t la = (uint32_t)(uintptr_t)&ldsB[buf][srow][schk];
        asm volatile("global_load_async_to_lds_b128 %0, %1, off"
                     :: "v"(la), "v"(ga) : "memory");
    };

    const int nK = K >> 5;
    stage(0, 0);

    v8f acc[4] = {};
    const __bf16* arow = A + (size_t)(m0 + lh) * K;

    for (int kt = 0; kt < nK; ++kt) {
        if (kt + 1 < nK) {                      // uniform branch
            stage(kt + 1, (kt + 1) & 1);        // next tile in flight...
            asm volatile("s_wait_asynccnt 1" ::: "memory");  // ...current complete
        } else {
            asm volatile("s_wait_asynccnt 0" ::: "memory");
        }
        __syncthreads();                        // all waves' staged rows visible

        const int k = kt << 5;
        __builtin_prefetch(arow + k + 128, 0, 0);
        Frag a;
        a.h[0] = *(const v8bf*)(arow + k + sel * 8);
        a.h[1] = *(const v8bf*)(arow + k + 16 + sel * 8);
        const __bf16(*bufB)[32] = ldsB[kt & 1];
#pragma unroll
        for (int nt = 0; nt < 4; ++nt) {
            Frag bf;
            bf.h[0] = *(const v8bf*)&bufB[nt * 16 + lh][sel * 8];
            bf.h[1] = *(const v8bf*)&bufB[nt * 16 + lh][16 + sel * 8];
            acc[nt] = __builtin_amdgcn_wmma_f32_16x16x32_bf16(
                false, a.v, false, bf.v, (short)0, acc[nt], false, false);
        }
        __syncthreads();                        // reads done before tile is re-staged
    }

#pragma unroll
    for (int nt = 0; nt < 4; ++nt) {
        const int n  = n0 + nt * 16 + lh;
        const float bv = bias[n];
#pragma unroll
        for (int r = 0; r < 8; ++r) {
            const int m = m0 + r + sel * 8;                 // C-frag row mapping
            C[(size_t)m * N + n] = acc[nt][r] + bv;
        }
    }
}

// ---------------------------------------------------------------------------
// RoPE on Q: q_lin (B*S, HQ*HD) -> qb bf16 laid out (B,HQ,S,HD)
// ---------------------------------------------------------------------------
__global__ void rope_q_kernel(const float* __restrict__ qlin,
                              const float* __restrict__ cosT,
                              const float* __restrict__ sinT,
                              __bf16* __restrict__ qb) {
    int idx = blockIdx.x * blockDim.x + threadIdx.x;    // 2^22 threads
    if (idx >= Bc * Sc * HQc * HDc) return;
    int d = idx & 63;
    int h = (idx >> 6) & 15;
    int s = (idx >> 10) & 2047;
    int b = idx >> 21;
    size_t base = ((size_t)(b * Sc + s)) * Dc + h * HDc;
    float v  = qlin[base + d];
    float vp = qlin[base + ((d < 32) ? d + 32 : d - 32)];
    float rot = (d < 32) ? -vp : vp;
    float o = v * cosT[s * HDc + d] + rot * sinT[s * HDc + d];
    qb[(((size_t)(b * HQc + h)) * Sc + s) * HDc + d] = (__bf16)o;
}

// RoPE on K: also writes present_k (fp32) into d_out
__global__ void rope_k_kernel(const float* __restrict__ klin,
                              const float* __restrict__ cosT,
                              const float* __restrict__ sinT,
                              float* __restrict__ presentK,
                              __bf16* __restrict__ kb) {
    int idx = blockIdx.x * blockDim.x + threadIdx.x;    // 2^20 threads
    if (idx >= Bc * Sc * HKVc * HDc) return;
    int d = idx & 63;
    int h = (idx >> 6) & 3;
    int s = (idx >> 8) & 2047;
    int b = idx >> 19;
    size_t base = ((size_t)(b * Sc + s)) * (HKVc * HDc) + h * HDc;
    float v  = klin[base + d];
    float vp = klin[base + ((d < 32) ? d + 32 : d - 32)];
    float rot = (d < 32) ? -vp : vp;
    float o = v * cosT[s * HDc + d] + rot * sinT[s * HDc + d];
    size_t oi = (((size_t)(b * HKVc + h)) * Sc + s) * HDc + d;
    presentK[oi] = o;
    kb[oi] = (__bf16)o;
}

// V prep: present_v (fp32, (B,HKV,S,HD)) + V^T bf16 (B,HKV,HD,S) for B-frags
__global__ void v_prep_kernel(const float* __restrict__ vlin,
                              float* __restrict__ presentV,
                              __bf16* __restrict__ vT) {
    int idx = blockIdx.x * blockDim.x + threadIdx.x;    // 2^20 threads
    if (idx >= Bc * Sc * HKVc * HDc) return;
    int d = idx & 63;
    int h = (idx >> 6) & 3;
    int s = (idx >> 8) & 2047;
    int b = idx >> 19;
    float v = vlin[((size_t)(b * Sc + s)) * (HKVc * HDc) + h * HDc + d];
    presentV[(((size_t)(b * HKVc + h)) * Sc + s) * HDc + d] = v;
    vT[(((size_t)(b * HKVc + h)) * HDc + d) * Sc + s] = (__bf16)v;
}

// ---------------------------------------------------------------------------
// Flash-style attention: one wave owns one 16-query tile of one head.
// Streams keys in chunks of 32: 4 WMMAs (Q.K^T) + online softmax + 4 WMMAs (P.V)
// grid = B*HQ*(S/16)/8 blocks of 256 threads.
// ---------------------------------------------------------------------------
__global__ void attention_kernel(const __bf16* __restrict__ qb,
                                 const __bf16* __restrict__ kb,
                                 const __bf16* __restrict__ vT,
                                 const int*    __restrict__ mask,
                                 __bf16* __restrict__ attnb) {
    __shared__ __align__(16) __bf16 ldsP[8][16][32];   // per-wave P bounce (8 KB)

    const int wave = threadIdx.x >> 5;
    const int lane = threadIdx.x & 31;
    const int lh   = lane & 15;
    const int sel  = lane >> 4;

    const int tilesPerHead = Sc / 16;
    const int tile = blockIdx.x * 8 + wave;
    const int b  = tile / (HQc * tilesPerHead);
    const int rm = tile % (HQc * tilesPerHead);
    const int h  = rm / tilesPerHead;
    const int q0 = (rm % tilesPerHead) * 16;
    const int hk = h / (HQc / HKVc);

    // Q fragments for d=[0,32) and d=[32,64)
    const __bf16* qrow = qb + (((size_t)(b * HQc + h)) * Sc + q0 + lh) * HDc;
    Frag qa0, qa1;
    qa0.h[0] = *(const v8bf*)(qrow + sel * 8);
    qa0.h[1] = *(const v8bf*)(qrow + 16 + sel * 8);
    qa1.h[0] = *(const v8bf*)(qrow + 32 + sel * 8);
    qa1.h[1] = *(const v8bf*)(qrow + 48 + sel * 8);

    const __bf16* kbase = kb + ((size_t)(b * HKVc + hk)) * Sc * HDc;
    const __bf16* vbase = vT + ((size_t)(b * HKVc + hk)) * HDc * Sc;
    const int*    mrow  = mask + b * Sc;

    float rowmax[8], rowsum[8];
#pragma unroll
    for (int r = 0; r < 8; ++r) { rowmax[r] = -1e30f; rowsum[r] = 0.f; }
    v8f acc[4] = {};

    for (int k0 = 0; k0 < Sc; k0 += 32) {
        // ---- scores for 32 keys: two 16x16 C-frags -------------------------
        v8f s0 = {}, s1 = {};
        {
            Frag kf;
            const __bf16* kr = kbase + (size_t)(k0 + lh) * HDc;
            kf.h[0] = *(const v8bf*)(kr + sel * 8);
            kf.h[1] = *(const v8bf*)(kr + 16 + sel * 8);
            s0 = __builtin_amdgcn_wmma_f32_16x16x32_bf16(false, qa0.v, false, kf.v,
                                                         (short)0, s0, false, false);
            kf.h[0] = *(const v8bf*)(kr + 32 + sel * 8);
            kf.h[1] = *(const v8bf*)(kr + 48 + sel * 8);
            s0 = __builtin_amdgcn_wmma_f32_16x16x32_bf16(false, qa1.v, false, kf.v,
                                                         (short)0, s0, false, false);
            const __bf16* kr2 = kbase + (size_t)(k0 + 16 + lh) * HDc;
            kf.h[0] = *(const v8bf*)(kr2 + sel * 8);
            kf.h[1] = *(const v8bf*)(kr2 + 16 + sel * 8);
            s1 = __builtin_amdgcn_wmma_f32_16x16x32_bf16(false, qa0.v, false, kf.v,
                                                         (short)0, s1, false, false);
            kf.h[0] = *(const v8bf*)(kr2 + 32 + sel * 8);
            kf.h[1] = *(const v8bf*)(kr2 + 48 + sel * 8);
            s1 = __builtin_amdgcn_wmma_f32_16x16x32_bf16(false, qa1.v, false, kf.v,
                                                         (short)0, s1, false, false);
        }
        const float padA = (mrow[k0 + lh] == 0)      ? -1e9f : 0.f;
        const float padB = (mrow[k0 + 16 + lh] == 0) ? -1e9f : 0.f;

        // ---- online softmax (row r lives on this lane's half) --------------
#pragma unroll
        for (int r = 0; r < 8; ++r) {
            float a = s0[r] * SCALE_F + padA;
            float c = s1[r] * SCALE_F + padB;
            float cmax = redmax16(fmaxf(a, c));
            float nm   = fmaxf(rowmax[r], cmax);
            float corr = __expf(rowmax[r] - nm);
            rowmax[r]  = nm;
            float p0 = __expf(a - nm);
            float p1 = __expf(c - nm);
            rowsum[r] = rowsum[r] * corr + redsum16(p0 + p1);
            acc[0][r] *= corr; acc[1][r] *= corr;
            acc[2][r] *= corr; acc[3][r] *= corr;
            const int m = r + sel * 8;
            ldsP[wave][m][lh]      = (__bf16)p0;
            ldsP[wave][m][16 + lh] = (__bf16)p1;
        }
        __syncthreads();   // uniform trip count across all waves in the block

        // ---- P (A-frag) from LDS, multiply by V^T B-frags ------------------
        Frag pa;
        pa.h[0] = *(const v8bf*)&ldsP[wave][lh][sel * 8];
        pa.h[1] = *(const v8bf*)&ldsP[wave][lh][16 + sel * 8];
#pragma unroll
        for (int nt = 0; nt < 4; ++nt) {
            const __bf16* vr = vbase + (size_t)(nt * 16 + lh) * Sc + k0;
            Frag vf;
            vf.h[0] = *(const v8bf*)(vr + sel * 8);
            vf.h[1] = *(const v8bf*)(vr + 16 + sel * 8);
            acc[nt] = __builtin_amdgcn_wmma_f32_16x16x32_bf16(
                false, pa.v, false, vf.v, (short)0, acc[nt], false, false);
        }
        __syncthreads();
    }

    // ---- normalize, emit bf16 activation rows (b,s, h*64+d) ----------------
#pragma unroll
    for (int nt = 0; nt < 4; ++nt) {
#pragma unroll
        for (int r = 0; r < 8; ++r) {
            const int m = r + sel * 8;
            float o = acc[nt][r] / rowsum[r];
            attnb[((size_t)(b * Sc) + q0 + m) * Dc + h * HDc + nt * 16 + lh] = (__bf16)o;
        }
    }
}

// ---------------------------------------------------------------------------
// Host-side orchestration
// ---------------------------------------------------------------------------
extern "C" void kernel_launch(void* const* d_in, const int* in_sizes, int n_in,
                              void* d_out, int out_size, void* d_ws, size_t ws_size,
                              hipStream_t stream) {
    const float* x    = (const float*)d_in[0];
    const float* cosT = (const float*)d_in[1];
    const float* sinT = (const float*)d_in[2];
    const int*   mask = (const int*)  d_in[3];
    const float* Wq   = (const float*)d_in[4];
    const float* bq   = (const float*)d_in[5];
    const float* Wk   = (const float*)d_in[6];
    const float* bk   = (const float*)d_in[7];
    const float* Wv   = (const float*)d_in[8];
    const float* bv   = (const float*)d_in[9];
    const float* Wo   = (const float*)d_in[10];
    const float* bo   = (const float*)d_in[11];

    float* out      = (float*)d_out;                       // (B,S,D)
    float* presentK = out + (size_t)Bc * Sc * Dc;          // (B,HKV,S,HD)
    float* presentV = presentK + (size_t)Bc * HKVc * Sc * HDc;

    char*  ws  = (char*)d_ws;
    size_t off = 0;
    auto alloc = [&](size_t bytes) -> void* {
        void* p = ws + off;
        off = (off + bytes + 255) & ~(size_t)255;
        return p;
    };

    const size_t nX   = (size_t)MTOT * Dc;            // 4M
    const size_t nKV  = (size_t)Bc * HKVc * Sc * HDc; // 1M

    __bf16* xb   = (__bf16*)alloc(nX * 2);
    __bf16* WqT  = (__bf16*)alloc((size_t)Dc * Dc * 2);
    __bf16* WkT  = (__bf16*)alloc((size_t)(HKVc * HDc) * Dc * 2);
    __bf16* WvT  = (__bf16*)alloc((size_t)(HKVc * HDc) * Dc * 2);
    __bf16* WoT  = (__bf16*)alloc((size_t)Dc * Dc * 2);
    float*  qlin = (float*) alloc(nX * 4);
    float*  klin = (float*) alloc((size_t)MTOT * HKVc * HDc * 4);
    float*  vlin = (float*) alloc((size_t)MTOT * HKVc * HDc * 4);
    __bf16* qbb  = (__bf16*)alloc(nX * 2);
    __bf16* kbb  = (__bf16*)alloc(nKV * 2);
    __bf16* vTT  = (__bf16*)alloc(nKV * 2);
    __bf16* attnb = (__bf16*)qlin;   // reuse q_lin scratch after RoPE consumes it

    const int T = 256;

    // 1) bf16 conversions / weight transposes
    cvt_bf16_kernel<<<(int)(nX / T), T, 0, stream>>>(x, xb, (int)nX);
    transpose_cvt_kernel<<<(Dc * Dc) / T, T, 0, stream>>>(Wq, WqT, Dc, Dc);
    transpose_cvt_kernel<<<(Dc * HKVc * HDc) / T, T, 0, stream>>>(Wk, WkT, Dc, HKVc * HDc);
    transpose_cvt_kernel<<<(Dc * HKVc * HDc) / T, T, 0, stream>>>(Wv, WvT, Dc, HKVc * HDc);
    transpose_cvt_kernel<<<(Dc * Dc) / T, T, 0, stream>>>(Wo, WoT, Dc, Dc);

    // 2) QKV projections (WMMA GEMMs, async-LDS staged weights)
    gemm_bias_bf16_kernel<<<dim3(Dc / 64, MTOT / 128), T, 0, stream>>>(
        xb, WqT, bq, qlin, MTOT, Dc, Dc);
    gemm_bias_bf16_kernel<<<dim3((HKVc * HDc) / 64, MTOT / 128), T, 0, stream>>>(
        xb, WkT, bk, klin, MTOT, HKVc * HDc, Dc);
    gemm_bias_bf16_kernel<<<dim3((HKVc * HDc) / 64, MTOT / 128), T, 0, stream>>>(
        xb, WvT, bv, vlin, MTOT, HKVc * HDc, Dc);

    // 3) RoPE + cache emission + V transpose
    rope_q_kernel<<<(int)(nX / T), T, 0, stream>>>(qlin, cosT, sinT, qbb);
    rope_k_kernel<<<(int)(nKV / T), T, 0, stream>>>(klin, cosT, sinT, presentK, kbb);
    v_prep_kernel<<<(int)(nKV / T), T, 0, stream>>>(vlin, presentV, vTT);

    // 4) attention (one wave per 16-query tile)
    const int nTiles = Bc * HQc * (Sc / 16);   // 4096
    attention_kernel<<<nTiles / 8, T, 0, stream>>>(qbb, kbb, vTT, mask, attnb);

    // 5) output projection straight into d_out
    gemm_bias_bf16_kernel<<<dim3(Dc / 64, MTOT / 128), T, 0, stream>>>(
        attnb, WoT, bo, out, MTOT, Dc, Dc);
}
